// LinearRandomAttention_73693048865484
// MI455X (gfx1250) — matrix operand order repaired and checked by reading, not compiled, not run on hardware
//
#include <hip/hip_runtime.h>

#define DEV __device__ __forceinline__

typedef __bf16 bf16_t;
typedef bf16_t v16bf __attribute__((ext_vector_type(16)));
typedef float v8f __attribute__((ext_vector_type(8)));
typedef unsigned int v8u __attribute__((ext_vector_type(8)));
typedef unsigned int v4u __attribute__((ext_vector_type(4)));
typedef unsigned short u16;

static constexpr int B_ = 4, N_ = 4096, HSZ_ = 1024, H_ = 16, D_ = 64, NB_ = 64, R_ = 3;
static constexpr float NEGM = -1e4f;
static constexpr float LN_EPS = 1e-12f;

DEV u16 f2bf(float f) {
  unsigned u = __builtin_bit_cast(unsigned, f);
  u += 0x7fffu + ((u >> 16) & 1u);
  return (u16)(u >> 16);
}

// ---- WMMA fragment loaders (CDNA5 16x16x32 bf16 layouts, wave32) ----
// A (16x32, MxK): lanes 0-15 row M=lane, elems {K0..K0+7, K0+16..K0+23};
//                 lanes 16-31 same rows, elems shifted by +8 in K.
DEV v16bf loadA(const u16* rp, int k0, int lane) {
  int off = (lane >> 4) << 3;  // 0 or 8
  v4u lo = *(const v4u*)(rp + k0 + off);
  v4u hi = *(const v4u*)(rp + k0 + 16 + off);
  v8u u;
  u[0] = lo[0]; u[1] = lo[1]; u[2] = lo[2]; u[3] = lo[3];
  u[4] = hi[0]; u[5] = hi[1]; u[6] = hi[2]; u[7] = hi[3];
  return __builtin_bit_cast(v16bf, u);
}
// B (32x16, KxN): lane holds column N=lane&15; lanes 0-15 K {0..7,16..23}, lanes 16-31 K {8..15,24..31}.
// np_ = pointer to this lane's column stored N-major (K contiguous, leading dim = row span).
DEV v16bf loadB(const u16* np_, int k0, int lane) {
  int off = (lane >> 4) << 3;  // 0 or 8
  v4u lo = *(const v4u*)(np_ + k0 + off);
  v4u hi = *(const v4u*)(np_ + k0 + 16 + off);
  v8u u;
  u[0] = lo[0]; u[1] = lo[1]; u[2] = lo[2]; u[3] = lo[3];
  u[4] = hi[0]; u[5] = hi[1]; u[6] = hi[2]; u[7] = hi[3];
  return __builtin_bit_cast(v16bf, u);
}

DEV v8f wmma_bf16(v16bf a, v16bf b, v8f c) {
  return __builtin_amdgcn_wmma_f32_16x16x32_bf16(false, a, false, b, (short)0, c, false, false);
}

// gfx1250 async global->LDS copy (ASYNCcnt-tracked). Generic shared pointers on
// amdgcn carry the LDS byte offset in their low 32 bits (aperture in [63:32]).
DEV void async_g2l_b128(u16* lds_dst, const u16* gsrc) {
  unsigned l = (unsigned)(size_t)lds_dst;
  unsigned long long g = (unsigned long long)(size_t)gsrc;
  asm volatile("global_load_async_to_lds_b128 %0, %1, off" ::"v"(l), "v"(g) : "memory");
}

// ---------------- conversion kernels ----------------
__global__ void __launch_bounds__(256) cast_bf16_k(const float* __restrict__ in,
                                                   u16* __restrict__ out, int n) {
  int i = blockIdx.x * 256 + threadIdx.x;
  if (i < n) out[i] = f2bf(in[i]);
}

__global__ void __launch_bounds__(256) transpose_bf16_k(const float* __restrict__ W,
                                                        u16* __restrict__ Wt) {
  int i = blockIdx.x * 256 + threadIdx.x;  // over HSZ*HSZ
  int k = i >> 10, n = i & (HSZ_ - 1);
  Wt[(size_t)n * HSZ_ + k] = f2bf(W[(size_t)k * HSZ_ + n]);
}

// ---------------- shared GEMM mainloop ----------------
// Block: 128 threads = 4 waves. Block tile 128(M) x 64(N). Wave tile 32 x 64.
// B panel (64 cols x 32 k = 4KB bf16) double-buffered in LDS via async loads.
DEV void issue_panel(const u16* __restrict__ Wt, int n0, int k0, u16* buf, int tid) {
#pragma unroll
  for (int j = 0; j < 2; ++j) {
    int c = tid + j * 128;          // 256 chunks of 8 u16
    int col = c >> 2, q = c & 3;    // col 0..63, quarter of 32-k
    const u16* g = Wt + (size_t)(n0 + col) * HSZ_ + k0 + q * 8;
    async_g2l_b128(buf + col * 32 + q * 8, g);
  }
}

DEV void gemm_mainloop(const u16* __restrict__ Abase, const u16* __restrict__ Wt,
                       int m0, int n0, int tid, int lane, u16* Blds, v8f acc[2][4]) {
  int c0 = lane & 15;
  const v8f zero = {0, 0, 0, 0, 0, 0, 0, 0};
#pragma unroll
  for (int mt = 0; mt < 2; ++mt)
#pragma unroll
    for (int t = 0; t < 4; ++t) acc[mt][t] = zero;

  const u16* arow0 = Abase + (size_t)(m0 + c0) * HSZ_;
  const u16* arow1 = arow0 + (size_t)16 * HSZ_;

  constexpr int NK = HSZ_ / 32;  // 32 k-steps
  issue_panel(Wt, n0, 0, Blds, tid);
  for (int kc = 0; kc < NK; ++kc) {
    u16* cur = Blds + (kc & 1) * 2048;
    if (kc + 1 < NK) {
      issue_panel(Wt, n0, (kc + 1) * 32, Blds + ((kc + 1) & 1) * 2048, tid);
      asm volatile("s_wait_asynccnt 2" ::: "memory");  // retire older panel only
    } else {
      asm volatile("s_wait_asynccnt 0" ::: "memory");
    }
    __syncthreads();  // cur panel visible to all waves
    int k0 = kc * 32;
    v16bf a0 = loadA(arow0, k0, lane);
    v16bf a1 = loadA(arow1, k0, lane);
    v16bf bfrag[4];
#pragma unroll
    for (int t = 0; t < 4; ++t) bfrag[t] = loadB(cur + (t * 16 + c0) * 32, 0, lane);
#pragma unroll
    for (int t = 0; t < 4; ++t) {
      acc[0][t] = wmma_bf16(a0, bfrag[t], acc[0][t]);
      acc[1][t] = wmma_bf16(a1, bfrag[t], acc[1][t]);
    }
    __syncthreads();  // done reading cur before it is overwritten
  }
}

// ---------------- QKV projection GEMM ----------------
// mode 0: Q (b,H,n,d) scaled by 0.125   mode 1: K (b,H,n,d)
// mode 2: V block-transposed (b,H,nb,d,64)
__global__ void __launch_bounds__(128) gemm_qkv_k(const u16* __restrict__ Xb,
                                                  const u16* __restrict__ Wt,
                                                  const float* __restrict__ bias,
                                                  u16* __restrict__ Out, int mode) {
  __shared__ __align__(16) u16 Blds[2 * 2048];
  int tid = threadIdx.x, lane = tid & 31, wave = tid >> 5;
  int half = lane >> 4, c0 = lane & 15;
  int m0 = blockIdx.x * 128 + wave * 32;
  int n0 = blockIdx.y * 64;
  v8f acc[2][4];
  gemm_mainloop(Xb, Wt, m0, n0, tid, lane, Blds, acc);

#pragma unroll
  for (int t = 0; t < 4; ++t) {
    int nc = n0 + t * 16 + c0;
    int hh = nc >> 6, dd = nc & 63;
    float bv = bias[nc];
#pragma unroll
    for (int mt = 0; mt < 2; ++mt)
#pragma unroll
      for (int v = 0; v < 8; ++v) {
        int m = m0 + mt * 16 + half * 8 + v;
        int bidx = m >> 12, n = m & (N_ - 1);
        float val = acc[mt][t][v] + bv;
        size_t addr;
        if (mode == 2) {
          int blk = n >> 6, kk = n & 63;
          addr = ((((size_t)bidx * H_ + hh) * NB_ + blk) * D_ + dd) * 64 + kk;
        } else {
          if (mode == 0) val *= 0.125f;  // 1/sqrt(64)
          addr = (((size_t)bidx * H_ + hh) * N_ + n) * D_ + dd;
        }
        Out[addr] = f2bf(val);
      }
  }
}

// ---------------- output projection + residual ----------------
__global__ void __launch_bounds__(128) gemm_out_k(const u16* __restrict__ Ctxb,
                                                  const u16* __restrict__ Wot,
                                                  const float* __restrict__ bo,
                                                  const float* __restrict__ xs,
                                                  float* __restrict__ y) {
  __shared__ __align__(16) u16 Blds[2 * 2048];
  int tid = threadIdx.x, lane = tid & 31, wave = tid >> 5;
  int half = lane >> 4, c0 = lane & 15;
  int m0 = blockIdx.x * 128 + wave * 32;
  int n0 = blockIdx.y * 64;
  v8f acc[2][4];
  gemm_mainloop(Ctxb, Wot, m0, n0, tid, lane, Blds, acc);

#pragma unroll
  for (int t = 0; t < 4; ++t) {
    int nc = n0 + t * 16 + c0;
    float bv = bo[nc];
#pragma unroll
    for (int mt = 0; mt < 2; ++mt)
#pragma unroll
      for (int v = 0; v < 8; ++v) {
        int m = m0 + mt * 16 + half * 8 + v;
        size_t idx = (size_t)m * HSZ_ + nc;
        y[idx] = acc[mt][t][v] + bv + xs[idx];
      }
  }
}

// ---------------- BigBird sparse attention (flash-style) ----------------
__global__ void __launch_bounds__(128) bigbird_attn_k(const u16* __restrict__ Qb,
                                                      const u16* __restrict__ Kb,
                                                      const u16* __restrict__ Vt,
                                                      const int* __restrict__ mask,
                                                      const int* __restrict__ rand_idx,
                                                      u16* __restrict__ Ctxb) {
  __shared__ __align__(16) u16 Plds[4 * 16 * 64];
  int qi = blockIdx.x, h = blockIdx.y, b = blockIdx.z;
  int tid = threadIdx.x, lane = tid & 31, wave = tid >> 5;
  int half = lane >> 4, c0 = lane & 15;
  int bh = b * H_ + h;

  int list[8];
  int nblk = 0;
  bool grow = (qi == 0) || (qi == NB_ - 1);
  if (!grow) {
    if (qi == 1) {
      list[0] = 0; list[1] = 1; list[2] = 2; list[3] = NB_ - 1; nblk = 4;
    } else if (qi == NB_ - 2) {
      list[0] = NB_ - 3; list[1] = NB_ - 2; list[2] = NB_ - 1; list[3] = 0; nblk = 4;
    } else {
      list[0] = qi - 1; list[1] = qi; list[2] = qi + 1; list[3] = 0; list[4] = NB_ - 1; nblk = 5;
    }
    int rrow = qi - 1;  // second uses row 0, penult row NB-3, middles row qi-1
#pragma unroll
    for (int j = 0; j < R_; ++j) list[nblk++] = rand_idx[(h * (NB_ - 2) + rrow) * R_ + j];
  }
  int total = grow ? NB_ : nblk;

  const u16* qrow = Qb + ((size_t)bh * N_ + qi * 64 + wave * 16 + c0) * D_;

  float mrow[8], lrow[8];
  const v8f zero = {0, 0, 0, 0, 0, 0, 0, 0};
  v8f cacc[4];
#pragma unroll
  for (int v = 0; v < 8; ++v) { mrow[v] = -1e30f; lrow[v] = 0.f; }
#pragma unroll
  for (int t = 0; t < 4; ++t) cacc[t] = zero;

  for (int it = 0; it < total; ++it) {
    int kb = grow ? it : list[it];
    // ---- scores: Qtile(16x64) @ Kblk^T(64x64) ----
    v8f s[4];
#pragma unroll
    for (int t = 0; t < 4; ++t) s[t] = zero;
    for (int ks = 0; ks < D_; ks += 32) {
      v16bf a = loadA(qrow, ks, lane);
      v16bf bm[4];
#pragma unroll
      for (int t = 0; t < 4; ++t) {
        const u16* kp = Kb + ((size_t)bh * N_ + kb * 64 + t * 16 + c0) * D_;
        bm[t] = loadB(kp, ks, lane);
      }
#pragma unroll
      for (int t = 0; t < 4; ++t) s[t] = wmma_bf16(a, bm[t], s[t]);
    }
    // additive key mask
    float cm[4];
#pragma unroll
    for (int t = 0; t < 4; ++t)
      cm[t] = (1.0f - (float)mask[b * N_ + kb * 64 + t * 16 + c0]) * NEGM;
    // ---- online softmax (per C-layout row v + 8*half) ----
#pragma unroll
    for (int v = 0; v < 8; ++v) {
      float mx = -1e30f;
#pragma unroll
      for (int t = 0; t < 4; ++t) {
        float x = s[t][v] + cm[t];
        s[t][v] = x;
        mx = fmaxf(mx, x);
      }
#pragma unroll
      for (int w = 1; w < 16; w <<= 1) mx = fmaxf(mx, __shfl_xor(mx, w, 32));
      float mn = fmaxf(mrow[v], mx);
      float corr = __expf(mrow[v] - mn);
      float sum = 0.f;
#pragma unroll
      for (int t = 0; t < 4; ++t) {
        float p = __expf(s[t][v] - mn);
        s[t][v] = p;
        sum += p;
      }
#pragma unroll
      for (int w = 1; w < 16; w <<= 1) sum += __shfl_xor(sum, w, 32);
      lrow[v] = lrow[v] * corr + sum;
      mrow[v] = mn;
#pragma unroll
      for (int t = 0; t < 4; ++t) cacc[t][v] *= corr;
    }
    // ---- P (C-layout) -> LDS row-major bf16 (per-wave region; DS in-order) ----
    u16* pw = Plds + wave * 1024;
#pragma unroll
    for (int v = 0; v < 8; ++v)
#pragma unroll
      for (int t = 0; t < 4; ++t)
        pw[(half * 8 + v) * 64 + t * 16 + c0] = f2bf(s[t][v]);
    asm volatile("s_wait_dscnt 0" ::: "memory");
    // ---- ctx += P(16x64) @ Vblk(64x64), V stored d-major ----
    const u16* prow = Plds + wave * 1024 + c0 * 64;
    const u16* vbase = Vt + (((size_t)bh * NB_ + kb) * D_) * 64;
    for (int ks = 0; ks < 64; ks += 32) {
      v16bf ap = loadA(prow, ks, lane);
      v16bf bv[4];
#pragma unroll
      for (int dt = 0; dt < 4; ++dt) bv[dt] = loadB(vbase + (dt * 16 + c0) * 64, ks, lane);
#pragma unroll
      for (int dt = 0; dt < 4; ++dt) cacc[dt] = wmma_bf16(ap, bv[dt], cacc[dt]);
    }
  }
  // ---- normalize, apply query mask, store ctx (b,n,hsz) bf16 ----
#pragma unroll
  for (int v = 0; v < 8; ++v) {
    int row = qi * 64 + wave * 16 + half * 8 + v;
    float inv = lrow[v] > 0.f ? 1.0f / lrow[v] : 0.f;
    float sc = inv * (float)mask[b * N_ + row];
#pragma unroll
    for (int dt = 0; dt < 4; ++dt)
      Ctxb[((size_t)(b * N_ + row)) * HSZ_ + h * D_ + dt * 16 + c0] = f2bf(cacc[dt][v] * sc);
  }
}

// ---------------- in-place LayerNorm over last dim (1024) ----------------
__global__ void __launch_bounds__(256) ln_k(float* __restrict__ y,
                                            const float* __restrict__ gamma,
                                            const float* __restrict__ beta) {
  __shared__ float sb[256], sb2[256];
  int tid = threadIdx.x;
  float* p = y + (size_t)blockIdx.x * HSZ_;
  float vals[4];
  float s = 0.f, s2 = 0.f;
#pragma unroll
  for (int j = 0; j < 4; ++j) {
    float x = p[tid + j * 256];
    vals[j] = x;
    s += x;
    s2 += x * x;
  }
  sb[tid] = s;
  sb2[tid] = s2;
  __syncthreads();
  for (int off = 128; off > 0; off >>= 1) {
    if (tid < off) { sb[tid] += sb[tid + off]; sb2[tid] += sb2[tid + off]; }
    __syncthreads();
  }
  float mu = sb[0] * (1.0f / HSZ_);
  float var = sb2[0] * (1.0f / HSZ_) - mu * mu;
  float inv = rsqrtf(var + LN_EPS);
#pragma unroll
  for (int j = 0; j < 4; ++j) {
    int c = tid + j * 256;
    p[c] = (vals[j] - mu) * inv * gamma[c] + beta[c];
  }
}

extern "C" void kernel_launch(void* const* d_in, const int* in_sizes, int n_in,
                              void* d_out, int out_size, void* d_ws, size_t ws_size,
                              hipStream_t stream) {
  const float* xs   = (const float*)d_in[0];
  const int*   mask = (const int*)d_in[1];
  const float* Wq = (const float*)d_in[2];  const float* bq = (const float*)d_in[3];
  const float* Wk = (const float*)d_in[4];  const float* bk = (const float*)d_in[5];
  const float* Wv = (const float*)d_in[6];  const float* bv = (const float*)d_in[7];
  const float* Wo = (const float*)d_in[8];  const float* bo = (const float*)d_in[9];
  const float* gamma = (const float*)d_in[10];
  const float* beta  = (const float*)d_in[11];
  const int* rand_idx = (const int*)d_in[12];

  const size_t BN = (size_t)B_ * N_;          // 16384
  const size_t XE = BN * HSZ_;                // 16,777,216 elems
  const size_t WE = (size_t)HSZ_ * HSZ_;      // 1,048,576 elems

  char* ws = (char*)d_ws;
  size_t o = 0;
  u16* Xb  = (u16*)(ws + o); o += XE * 2;
  u16* Wqt = (u16*)(ws + o); o += WE * 2;
  u16* Wkt = (u16*)(ws + o); o += WE * 2;
  u16* Wvt = (u16*)(ws + o); o += WE * 2;
  u16* Wot = (u16*)(ws + o); o += WE * 2;
  u16* Qb  = (u16*)(ws + o); o += XE * 2;
  u16* Kb  = (u16*)(ws + o); o += XE * 2;
  u16* Vt  = (u16*)(ws + o); o += XE * 2;
  u16* Ctx = (u16*)(ws + o); o += XE * 2;
  float* y = (float*)d_out;

  cast_bf16_k<<<(int)((XE + 255) / 256), 256, 0, stream>>>(xs, Xb, (int)XE);
  transpose_bf16_k<<<(int)(WE / 256), 256, 0, stream>>>(Wq, Wqt);
  transpose_bf16_k<<<(int)(WE / 256), 256, 0, stream>>>(Wk, Wkt);
  transpose_bf16_k<<<(int)(WE / 256), 256, 0, stream>>>(Wv, Wvt);
  transpose_bf16_k<<<(int)(WE / 256), 256, 0, stream>>>(Wo, Wot);

  dim3 gg((unsigned)(BN / 128), HSZ_ / 64);  // (128,16)
  gemm_qkv_k<<<gg, 128, 0, stream>>>(Xb, Wqt, bq, Qb, 0);
  gemm_qkv_k<<<gg, 128, 0, stream>>>(Xb, Wkt, bk, Kb, 1);
  gemm_qkv_k<<<gg, 128, 0, stream>>>(Xb, Wvt, bv, Vt, 2);

  bigbird_attn_k<<<dim3(NB_, H_, B_), 128, 0, stream>>>(Qb, Kb, Vt, mask, rand_idx, Ctx);

  gemm_out_k<<<gg, 128, 0, stream>>>(Ctx, Wot, bo, xs, y);
  ln_k<<<(int)BN, 256, 0, stream>>>(y, gamma, beta);
}
